// Transformer_36653250904264
// MI455X (gfx1250) — matrix operand order repaired and checked by reading, not compiled, not run on hardware
//
#include <hip/hip_runtime.h>
#include <hip/hip_bf16.h>

typedef _Float16 v16h __attribute__((ext_vector_type(16)));
typedef _Float16 v8h  __attribute__((ext_vector_type(8)));
typedef _Float16 v4h  __attribute__((ext_vector_type(4)));
typedef float    v8f  __attribute__((ext_vector_type(8)));
typedef int      v4i  __attribute__((ext_vector_type(4)));

#define NTOK   16384     // tokens per batch image (128*128)
#define DIM    128
#define NHEADS 4
#define HD     32
#define MLPD   512
#define GW     128       // grid width = height
#define EPS    1e-3f

#define AS1 __attribute__((address_space(1)))
#define AS3 __attribute__((address_space(3)))

// CDNA5 async global->LDS path (direct-to-LDS, tracked by ASYNCcnt).
// Gated on builtin availability; falls back to VGPR staging otherwise.
#if defined(__has_builtin)
#  if __has_builtin(__builtin_amdgcn_global_load_async_to_lds_b128) && \
      __has_builtin(__builtin_amdgcn_s_wait_asynccnt)
#    define USE_ASYNC_LDS 1
#  endif
#endif
#ifndef USE_ASYNC_LDS
#  define USE_ASYNC_LDS 0
#endif

// ---------------------------------------------------------------------------
// Weight convert + transpose: fp32 [K][N] row-major  ->  f16 [N][K] row-major
// ---------------------------------------------------------------------------
__global__ void convert_wt_kernel(const float* __restrict__ W,
                                  _Float16* __restrict__ Wt, int K, int N) {
    int idx = blockIdx.x * 256 + threadIdx.x;
    if (idx >= K * N) return;
    int k = idx / N, n = idx % N;
    Wt[(size_t)n * K + k] = (_Float16)W[idx];
}

// ---------------------------------------------------------------------------
// LayerNorm: one wave32 per row of 128 fp32; writes f16 row (A of next GEMM)
// ---------------------------------------------------------------------------
__global__ void ln_kernel(const float* __restrict__ x,
                          const float* __restrict__ g,
                          const float* __restrict__ b,
                          _Float16* __restrict__ y) {
    int wid  = (blockIdx.x * blockDim.x + threadIdx.x) >> 5;   // row
    int lane = threadIdx.x & 31;
    const float4 xv = *(const float4*)(x + (size_t)wid * DIM + lane * 4);
    float s = xv.x + xv.y + xv.z + xv.w;
    float q = xv.x * xv.x + xv.y * xv.y + xv.z * xv.z + xv.w * xv.w;
    #pragma unroll
    for (int m = 16; m; m >>= 1) {
        s += __shfl_xor(s, m, 32);
        q += __shfl_xor(q, m, 32);
    }
    float mean = s * (1.f / DIM);
    float var  = q * (1.f / DIM) - mean * mean;
    float inv  = rsqrtf(var + EPS);
    const float4 gv = *(const float4*)(g + lane * 4);
    const float4 bv = *(const float4*)(b + lane * 4);
    v4h o;
    o[0] = (_Float16)((xv.x - mean) * inv * gv.x + bv.x);
    o[1] = (_Float16)((xv.y - mean) * inv * gv.y + bv.y);
    o[2] = (_Float16)((xv.z - mean) * inv * gv.z + bv.z);
    o[3] = (_Float16)((xv.w - mean) * inv * gv.w + bv.w);
    *(v4h*)(y + (size_t)wid * DIM + lane * 4) = o;
}

// ---------------------------------------------------------------------------
// 3x3 neighborhood attention. qkv: f16 [NTOK][384] (q|k|v, heads of 32 chans).
// One wave per (pos, head); lane == channel. Zero-padded neighbors give
// dot==0 and v==0, matching the reference's pad-then-softmax exactly.
// ---------------------------------------------------------------------------
__global__ void attn_kernel(const _Float16* __restrict__ qkv,
                            _Float16* __restrict__ o) {
    int w    = (blockIdx.x * blockDim.x + threadIdx.x) >> 5;   // 0..65535
    int lane = threadIdx.x & 31;                               // channel
    int pos  = w >> 2;
    int head = w & 3;
    int i = pos >> 7, j = pos & (GW - 1);
    const size_t base = (size_t)pos * 384 + head * HD + lane;
    const float qc = (float)qkv[base];

    float dots[9], vv[9];
    #pragma unroll
    for (int nb = 0; nb < 9; ++nb) {
        int di = nb / 3 - 1, dj = nb % 3 - 1;
        int ii = i + di, jj = j + dj;
        bool ok = ((unsigned)ii < (unsigned)GW) && ((unsigned)jj < (unsigned)GW);
        size_t nbase = (size_t)(ii * GW + jj) * 384 + head * HD + lane;
        float kc = ok ? (float)qkv[nbase + 128] : 0.f;
        vv[nb]   = ok ? (float)qkv[nbase + 256] : 0.f;
        float d = qc * kc;
        #pragma unroll
        for (int m = 16; m; m >>= 1) d += __shfl_xor(d, m, 32);
        dots[nb] = d * 0.17677669529663687f;                   // HD^-0.5
    }
    float mx = dots[0];
    #pragma unroll
    for (int nb = 1; nb < 9; ++nb) mx = fmaxf(mx, dots[nb]);
    float den = 0.f, acc = 0.f;
    #pragma unroll
    for (int nb = 0; nb < 9; ++nb) {
        float e = __expf(dots[nb] - mx);
        den += e;
        acc += e * vv[nb];
    }
    o[(size_t)pos * DIM + head * HD + lane] = (_Float16)(acc / den);
}

// ---------------------------------------------------------------------------
// WMMA GEMM:  out = act(A[M,K]f16 @ Bt[N,K]^T f16 + bias) (+ resid)
// 256 threads = 8 waves (4 row x 2 col); block tile 128x64; wave 32x32
// (2x2 v_wmma_f32_16x16x32_f16); BK=32.
// A tile double-buffered in LDS, filled with GLOBAL_LOAD_ASYNC_TO_LDS_B128
// (ASYNCcnt-tracked) when available, so the next tile's fill overlaps the
// current tile's WMMAs; only the older 2 async ops are waited on per step.
// ---------------------------------------------------------------------------
template <bool OUT_F16, bool HAS_BIAS, bool DO_GELU, bool HAS_RESID>
__global__ void gemm_kernel(const _Float16* __restrict__ A,
                            const _Float16* __restrict__ Bt,
                            const float* __restrict__ bias,
                            const float* __restrict__ resid,
                            void* __restrict__ out, int N, int K) {
    __shared__ _Float16 aLds[2][128 * 40];    // padded stride 40 halves

    const int tid  = threadIdx.x;
    const int lane = tid & 31;
    const int wave = tid >> 5;
    const int wr   = wave & 3;          // wave row group (0..3)
    const int wc   = wave >> 2;         // wave col group (0..1)
    const int m0   = blockIdx.y * 128;
    const int n0   = blockIdx.x * 64;

    // Hint the weight tile into cache (lowers to global_prefetch_b8)
    if (tid < 64)
        __builtin_prefetch(Bt + (size_t)(n0 + tid) * K, 0, 1);

    const v8f vzero = {0.f, 0.f, 0.f, 0.f, 0.f, 0.f, 0.f, 0.f};
    v8f acc[2][2];
    acc[0][0] = vzero; acc[0][1] = vzero; acc[1][0] = vzero; acc[1][1] = vzero;

    const int ka  = (lane >> 4) << 3;   // A frag K base: 0 or 8
    const int kb  = (lane >> 4) << 3;   // B frag K base: 0 or 8
    const int l15 = lane & 15;

    // Per-thread A-tile chunk coordinates (2 chunks x 8 halves = 16B each)
    const int r0  = tid >> 2,          ko0 = (tid & 3) << 3;
    const int r1  = (tid + 256) >> 2,  ko1 = ((tid + 256) & 3) << 3;

    const int nsteps = K >> 5;

    auto issue_tile = [&](int step, int buf) {
        const _Float16* g0 = A + (size_t)(m0 + r0) * K + step * 32 + ko0;
        const _Float16* g1 = A + (size_t)(m0 + r1) * K + step * 32 + ko1;
#if USE_ASYNC_LDS
        __builtin_amdgcn_global_load_async_to_lds_b128(
            (AS1 v4i*)(AS1 void*)(void*)g0,
            (AS3 v4i*)(AS3 void*)(void*)&aLds[buf][r0 * 40 + ko0], 0, 0);
        __builtin_amdgcn_global_load_async_to_lds_b128(
            (AS1 v4i*)(AS1 void*)(void*)g1,
            (AS3 v4i*)(AS3 void*)(void*)&aLds[buf][r1 * 40 + ko1], 0, 0);
#else
        v8h a0 = *(const v8h*)g0;
        v8h a1 = *(const v8h*)g1;
        *(v8h*)(&aLds[buf][r0 * 40 + ko0]) = a0;
        *(v8h*)(&aLds[buf][r1 * 40 + ko1]) = a1;
#endif
    };

    issue_tile(0, 0);
    for (int s = 0; s < nsteps; ++s) {
        const int cur = s & 1;
        const bool has_next = (s + 1 < nsteps);
        if (has_next) issue_tile(s + 1, cur ^ 1);
#if USE_ASYNC_LDS
        if (has_next) __builtin_amdgcn_s_wait_asynccnt(2);  // current buf done
        else          __builtin_amdgcn_s_wait_asynccnt(0);
#endif
        __syncthreads();                 // tile[cur] visible to all waves

        const int kc = s * 32;
        // A fragments (ISA 16-bit A 16x32 layout)
        v16h afrag[2];
        #pragma unroll
        for (int tm = 0; tm < 2; ++tm) {
            const _Float16* ap = &aLds[cur][(wr * 32 + tm * 16 + l15) * 40 + ka];
            v8h lo = *(const v8h*)ap;            // K = ka .. ka+7
            v8h hi = *(const v8h*)(ap + 16);     // K = ka+16 .. ka+23
            afrag[tm] = __builtin_shufflevector(lo, hi, 0, 1, 2, 3, 4, 5, 6, 7,
                                                8, 9, 10, 11, 12, 13, 14, 15);
        }
        // B fragments straight from L1/L2-resident transposed weights
        v16h bfrag[2];
        #pragma unroll
        for (int tn = 0; tn < 2; ++tn) {
            const _Float16* bp =
                Bt + (size_t)(n0 + wc * 32 + tn * 16 + l15) * K + kc + kb;
            v8h lo = *(const v8h*)bp;            // K = kb .. kb+7
            v8h hi = *(const v8h*)(bp + 16);     // K = kb+16 .. kb+23
            bfrag[tn] = __builtin_shufflevector(lo, hi, 0, 1, 2, 3, 4, 5, 6, 7,
                                                8, 9, 10, 11, 12, 13, 14, 15);
        }
        #pragma unroll
        for (int tm = 0; tm < 2; ++tm)
            #pragma unroll
            for (int tn = 0; tn < 2; ++tn)
                acc[tm][tn] = __builtin_amdgcn_wmma_f32_16x16x32_f16(
                    false, afrag[tm], false, bfrag[tn], (short)0, acc[tm][tn],
                    false, false);

        __syncthreads();                 // all waves done reading tile[cur]
    }

    // Epilogue: C/D layout -> lane 0-15: N=lane, M=v; lane 16-31: N=lane-16, M=v+8
    const int mAdd = (lane >> 4) * 8;
    #pragma unroll
    for (int tm = 0; tm < 2; ++tm)
        #pragma unroll
        for (int tn = 0; tn < 2; ++tn)
            #pragma unroll
            for (int v = 0; v < 8; ++v) {
                int m = m0 + wr * 32 + tm * 16 + mAdd + v;
                int n = n0 + wc * 32 + tn * 16 + l15;
                float val = acc[tm][tn][v];
                if (HAS_BIAS) val += bias[n];
                if (DO_GELU)  val = 0.5f * val * (1.f + erff(val * 0.70710678118654752f));
                if (HAS_RESID) val += resid[(size_t)m * N + n];
                if (OUT_F16)
                    ((_Float16*)out)[(size_t)m * N + n] = (_Float16)val;
                else
                    ((float*)out)[(size_t)m * N + n] = val;
            }
}

// ---------------------------------------------------------------------------
extern "C" void kernel_launch(void* const* d_in, const int* in_sizes, int n_in,
                              void* d_out, int out_size, void* d_ws, size_t ws_size,
                              hipStream_t stream) {
    const float* ln1_g = (const float*)d_in[1];
    const float* ln1_b = (const float*)d_in[2];
    const float* qkv_w = (const float*)d_in[3];
    const float* out_w = (const float*)d_in[4];
    const float* out_b = (const float*)d_in[5];
    const float* ln2_g = (const float*)d_in[6];
    const float* ln2_b = (const float*)d_in[7];
    const float* ff_w1 = (const float*)d_in[8];
    const float* ff_b1 = (const float*)d_in[9];
    const float* ff_w2 = (const float*)d_in[10];
    const float* ff_b2 = (const float*)d_in[11];

    // Workspace layout (bytes)
    char* ws = (char*)d_ws;
    _Float16* WqkvT = (_Float16*)ws;                 ws += (size_t)4 * 384 * 128 * 2;
    _Float16* WoutT = (_Float16*)ws;                 ws += (size_t)4 * 128 * 128 * 2;
    _Float16* W1T   = (_Float16*)ws;                 ws += (size_t)4 * 512 * 128 * 2;
    _Float16* W2T   = (_Float16*)ws;                 ws += (size_t)4 * 128 * 512 * 2;
    _Float16* ybuf  = (_Float16*)ws;                 ws += (size_t)NTOK * DIM * 2;
    _Float16* obuf  = (_Float16*)ws;                 ws += (size_t)NTOK * DIM * 2;
    _Float16* qkvb  = (_Float16*)ws;                 ws += (size_t)NTOK * 384 * 2;
    _Float16* hbuf  = (_Float16*)ws;                 ws += (size_t)NTOK * MLPD * 2;

    // Convert + transpose all weights to f16 [N][K]
    for (int d = 0; d < 4; ++d) {
        convert_wt_kernel<<<(128 * 384 + 255) / 256, 256, 0, stream>>>(
            qkv_w + (size_t)d * 128 * 384, WqkvT + (size_t)d * 384 * 128, 128, 384);
        convert_wt_kernel<<<(128 * 128 + 255) / 256, 256, 0, stream>>>(
            out_w + (size_t)d * 128 * 128, WoutT + (size_t)d * 128 * 128, 128, 128);
        convert_wt_kernel<<<(128 * 512 + 255) / 256, 256, 0, stream>>>(
            ff_w1 + (size_t)d * 128 * 512, W1T + (size_t)d * 512 * 128, 128, 512);
        convert_wt_kernel<<<(512 * 128 + 255) / 256, 256, 0, stream>>>(
            ff_w2 + (size_t)d * 512 * 128, W2T + (size_t)d * 128 * 512, 512, 128);
    }

    // Running x lives in d_out (fp32), updated in-place layer by layer
    (void)hipMemcpyAsync(d_out, d_in[0], (size_t)8 * NTOK * DIM * sizeof(float),
                         hipMemcpyDeviceToDevice, stream);

    const dim3 blk(256);
    for (int d = 0; d < 4; ++d) {
        for (int bi = 0; bi < 8; ++bi) {
            float* xb = (float*)d_out + (size_t)bi * NTOK * DIM;

            ln_kernel<<<NTOK / 8, blk, 0, stream>>>(
                xb, ln1_g + d * DIM, ln1_b + d * DIM, ybuf);

            gemm_kernel<true, false, false, false>
                <<<dim3(384 / 64, NTOK / 128), blk, 0, stream>>>(
                    ybuf, WqkvT + (size_t)d * 384 * 128, nullptr, nullptr,
                    qkvb, 384, 128);

            attn_kernel<<<NTOK * NHEADS / 8, blk, 0, stream>>>(qkvb, obuf);

            gemm_kernel<false, true, false, true>
                <<<dim3(DIM / 64, NTOK / 128), blk, 0, stream>>>(
                    obuf, WoutT + (size_t)d * 128 * 128, out_b + d * DIM, xb,
                    xb, DIM, 128);

            ln_kernel<<<NTOK / 8, blk, 0, stream>>>(
                xb, ln2_g + d * DIM, ln2_b + d * DIM, ybuf);

            gemm_kernel<true, true, true, false>
                <<<dim3(MLPD / 64, NTOK / 128), blk, 0, stream>>>(
                    ybuf, W1T + (size_t)d * 512 * 128, ff_b1 + d * MLPD, nullptr,
                    hbuf, MLPD, 128);

            gemm_kernel<false, true, false, true>
                <<<dim3(DIM / 64, NTOK / 128), blk, 0, stream>>>(
                    hbuf, W2T + (size_t)d * 128 * 512, ff_b2 + d * DIM, xb,
                    xb, DIM, 512);
        }
    }
}
